// TestModel_52913997087011
// MI455X (gfx1250) — matrix operand order repaired and checked by reading, not compiled, not run on hardware
//
#include <hip/hip_runtime.h>

// ---------------------------------------------------------------------------
// ROI-Align recast as per-(box,channel) dense matmuls on CDNA5 WMMA:
//   P(64x64) = Wy(64x48) * F(48x48) * WxT(48x64)      (K padded to 64)
// f16 operands / f32 accum via v_wmma_f32_16x16x32_f16 (wave32).
// Output stream (256 MB) is the roofline limiter; math hides under it.
// ---------------------------------------------------------------------------

typedef __attribute__((ext_vector_type(16))) _Float16 v16h;
typedef __attribute__((ext_vector_type(8)))  _Float16 v8h;
typedef __attribute__((ext_vector_type(8)))  float    v8f;

#define HIMG   160
#define WIMG   160
#define CCH    256
#define KBOX   64
#define POOLED 64
#define WIN    48      // feature window always covers one ROI (span <= 42)
#define KP     64      // padded K stride (f16 elements) for WMMA k-steps

// ---- A operand, 16x32 f16 tile from a row-major LDS matrix (stride KP) ----
// ISA 7.12.2: lane<16 holds row (lane&15), K = {kb..kb+7, kb+16..kb+23}
//             lane>=16 holds same row,    K = {kb+8..kb+15, kb+24..kb+31}
__device__ inline v16h load_a(const _Float16* M, int mrow, int kb, int lane) {
    const int r    = lane & 15;
    const int koff = (lane < 16) ? 0 : 8;
    const _Float16* base = M + (mrow + r) * KP + kb + koff;
    v8h lo = *(const v8h*)(base);        // 16B -> ds_load_b128
    v8h hi = *(const v8h*)(base + 16);   // 16B -> ds_load_b128
    v16h out;
#pragma unroll
    for (int i = 0; i < 8; ++i) { out[i] = lo[i]; out[i + 8] = hi[i]; }
    return out;
}

// ---- B operand, 32x16 f16 tile from a column-major LDS matrix ------------
// Storage: element (k, n) at M[n*KP + k].
// lane<16 holds col (lane&15), K = {kb..kb+7, kb+16..kb+23} ; lane>=16 holds K = {kb+8..kb+15, kb+24..kb+31}
__device__ inline v16h load_b(const _Float16* M, int kb, int ncol, int lane) {
    const int n    = lane & 15;
    const int koff = (lane < 16) ? 0 : 8;
    const _Float16* base = M + (ncol + n) * KP + kb + koff;
    v8h lo = *(const v8h*)(base);
    v8h hi = *(const v8h*)(base + 16);
    v16h out;
#pragma unroll
    for (int i = 0; i < 8; ++i) { out[i] = lo[i]; out[i + 8] = hi[i]; }
    return out;
}

__global__ __launch_bounds__(64) void roi_align_wmma_kernel(
    const float* __restrict__ images,   // [2, 256, 160, 160]
    const float* __restrict__ rois,     // [64, 5]
    float* __restrict__ out)            // [64, 256, 64, 64]
{
    // Shared: Wy row-major [64][KP], WxT "B layout" [pw][KP] (x contiguous),
    // per-wave Fc column-major [x][KP] (y contiguous), per-wave U row-major.
    __shared__ _Float16 sWy [64 * KP];          // 8 KB
    __shared__ _Float16 sWxT[64 * KP];          // 8 KB
    __shared__ _Float16 sFc [2][WIN * KP];      // 12 KB
    __shared__ _Float16 sU  [2][64 * KP];       // 16 KB   (total 44 KB)

    const int k    = blockIdx.x;                // box
    const int cb   = blockIdx.y;                // channel pair
    const int tid  = threadIdx.x;
    const int wave = tid >> 5;
    const int lane = tid & 31;

    // ---------------- box parameters (uniform; every thread computes) -----
    const float scale = 160.0f / 640.0f;        // dynamic spatial scale
    const int   b   = (int)rois[k * 5 + 0];
    const float bx1 = rois[k * 5 + 1] * scale - 0.5f;
    const float by1 = rois[k * 5 + 2] * scale - 0.5f;
    const float bx2 = rois[k * 5 + 3] * scale - 0.5f;
    const float by2 = rois[k * 5 + 4] * scale - 0.5f;
    const float bin_w = (bx2 - bx1) * (1.0f / 64.0f);
    const float bin_h = (by2 - by1) * (1.0f / 64.0f);

    int y_org = (int)floorf(by1);
    y_org = (y_org < 0) ? 0 : ((y_org > HIMG - WIN) ? (HIMG - WIN) : y_org);
    int x_org = (int)floorf(bx1);
    x_org = (x_org < 0) ? 0 : ((x_org > WIMG - WIN) ? (WIMG - WIN) : x_org);

    // ---------------- interpolation weight matrices (thread tid = row) ----
    {
        const int r = tid;                      // ph row for Wy, pw row for WxT
        v8h z = {};
#pragma unroll
        for (int i = 0; i < KP; i += 8) {
            *(v8h*)(sWy  + r * KP + i) = z;
            *(v8h*)(sWxT + r * KP + i) = z;
        }
        // ---- Wy row: SR=2 samples, 2 bilinear taps each, averaged by 1/2 ----
        {
            float w[3] = {0.f, 0.f, 0.f};
            int i0 = 0;
#pragma unroll
            for (int s = 0; s < 2; ++s) {
                float yy = by1 + ((float)(r * 2 + s) + 0.5f) * 0.5f * bin_h;
                float yc = fminf(fmaxf(yy, 0.0f), (float)(HIMG - 1));
                int   yl = (int)floorf(yc);
                float ly = yc - (float)yl;
                int   yh = yl + 1; if (yh > HIMG - 1) yh = HIMG - 1;
                if (s == 0) i0 = yl;
                w[yl - i0] += 0.5f * (1.0f - ly);
                w[yh - i0] += 0.5f * ly;
            }
#pragma unroll
            for (int t = 0; t < 3; ++t)
                sWy[r * KP + (i0 - y_org) + t] = (_Float16)w[t];
        }
        // ---- Wx row (stored transposed: [pw][x], x contiguous) ----
        {
            float w[3] = {0.f, 0.f, 0.f};
            int i0 = 0;
#pragma unroll
            for (int s = 0; s < 2; ++s) {
                float xx = bx1 + ((float)(r * 2 + s) + 0.5f) * 0.5f * bin_w;
                float xc = fminf(fmaxf(xx, 0.0f), (float)(WIMG - 1));
                int   xl = (int)floorf(xc);
                float lx = xc - (float)xl;
                int   xh = xl + 1; if (xh > WIMG - 1) xh = WIMG - 1;
                if (s == 0) i0 = xl;
                w[xl - i0] += 0.5f * (1.0f - lx);
                w[xh - i0] += 0.5f * lx;
            }
#pragma unroll
            for (int t = 0; t < 3; ++t)
                sWxT[r * KP + (i0 - x_org) + t] = (_Float16)w[t];
        }
    }

    // ---------------- per-wave: stage feature window into LDS (f16) -------
    const int c = cb * 2 + wave;
    _Float16* Fc = sFc[wave];
    _Float16* U  = sU[wave];
    {
        v8h z = {};
        for (int i = lane * 8; i < WIN * KP; i += 32 * 8) *(v8h*)(Fc + i) = z;
        for (int i = lane * 8; i < 64 * KP;  i += 32 * 8) *(v8h*)(U  + i) = z;

        const float* src = images +
            ((((size_t)b * CCH + c) * HIMG + y_org) * WIMG + x_org);
        for (int idx = lane; idx < WIN * WIN; idx += 32) {
            int y = idx / WIN;
            int x = idx - y * WIN;
            Fc[x * KP + y] = (_Float16)src[y * WIMG + x];   // column-major
        }
    }
    __syncthreads();

    // ---------------- stage A: U(64x48) = Wy(64x64k) * Fc(64k x48) --------
#pragma unroll
    for (int mt = 0; mt < 4; ++mt) {
#pragma unroll
        for (int nt = 0; nt < 3; ++nt) {
            v8f acc = {};
#pragma unroll
            for (int kt = 0; kt < 2; ++kt) {
                v16h a  = load_a(sWy, mt * 16, kt * 32, lane);
                v16h bm = load_b(Fc,  kt * 32, nt * 16, lane);
                acc = __builtin_amdgcn_wmma_f32_16x16x32_f16(
                    false, a, false, bm, (short)0, acc, false, false);
            }
            const int row = mt * 16 + ((lane < 16) ? 0 : 8);
            const int col = nt * 16 + (lane & 15);
#pragma unroll
            for (int rr = 0; rr < 8; ++rr)
                U[(row + rr) * KP + col] = (_Float16)acc[rr];  // D -> A layout
        }
    }

    // ---------------- stage B: P(64x64) = U(64x64k) * WxT(64k x64) --------
    float* dst = out + (((size_t)k * CCH + c) * POOLED) * POOLED;
#pragma unroll
    for (int mt = 0; mt < 4; ++mt) {
#pragma unroll
        for (int nt = 0; nt < 4; ++nt) {
            v8f acc = {};
#pragma unroll
            for (int kt = 0; kt < 2; ++kt) {
                v16h a  = load_a(U,    mt * 16, kt * 32, lane);
                v16h bm = load_b(sWxT, kt * 32, nt * 16, lane);
                acc = __builtin_amdgcn_wmma_f32_16x16x32_f16(
                    false, a, false, bm, (short)0, acc, false, false);
            }
            const int row = mt * 16 + ((lane < 16) ? 0 : 8);
            const int col = nt * 16 + (lane & 15);
#pragma unroll
            for (int rr = 0; rr < 8; ++rr)
                dst[(row + rr) * POOLED + col] = acc[rr];  // coalesced 64B runs
        }
    }
}

extern "C" void kernel_launch(void* const* d_in, const int* in_sizes, int n_in,
                              void* d_out, int out_size, void* d_ws, size_t ws_size,
                              hipStream_t stream) {
    const float* images = (const float*)d_in[0];   // [2,256,160,160] fp32
    const float* rois   = (const float*)d_in[1];   // [64,5] fp32
    float* out = (float*)d_out;                    // [64,256,64,64] fp32

    dim3 grid(KBOX, CCH / 2);   // 64 boxes x 128 channel-pairs
    dim3 block(64);             // 2 waves (wave32), one channel each
    roi_align_wmma_kernel<<<grid, block, 0, stream>>>(images, rois, out);
}
